// PointnetFPModuleGAG_48215302865128
// MI455X (gfx1250) — matrix operand order repaired and checked by reading, not compiled, not run on hardware
//
#include <hip/hip_runtime.h>
#include <hip/hip_bf16.h>
#include <float.h>

typedef __attribute__((ext_vector_type(16))) _Float16 v16h;
typedef __attribute__((ext_vector_type(8)))  _Float16 v8h;
typedef __attribute__((ext_vector_type(8)))  float    v8f;

// Problem constants (from reference)
#define BATCH  4
#define NU     16384          // unknown points per batch
#define MK     4096           // known points per batch
#define COLS   (BATCH*NU)     // 65536 total query columns
#define C1IN   133
#define C2IN   261
#define NCOMP  5
#define CK     256            // kfeat channels
#define CU     128            // ufeat channels
#define K1DIM  384            // gemm1 K (256 interp + 128 ufeat)
#define K2DIM  256            // gemm2 K
#define CO     256            // conv output channels
#define SQRT10 3.16227766016838f
#define BN_EPS 1e-5f

// Workspace layout (bytes)
static constexpr size_t OFF_A16 = 0;                                      // f16 [COLS][8]
static constexpr size_t OFF_AA  = OFF_A16 + (size_t)COLS*8*2;             // f32 [COLS]
static constexpr size_t OFF_B16 = OFF_AA  + (size_t)COLS*4;               // f16 [B*MK][8]
static constexpr size_t OFF_BB  = OFF_B16 + (size_t)BATCH*MK*8*2;         // f32 [B*MK]
static constexpr size_t OFF_IDX = OFF_BB  + (size_t)BATCH*MK*4;           // i32 [COLS][3]
static constexpr size_t OFF_WGT = OFF_IDX + (size_t)COLS*3*4;             // f32 [COLS][3]
static constexpr size_t OFF_KT  = OFF_WGT + (size_t)COLS*3*4;             // f16 [B*MK][256]
static constexpr size_t OFF_W1H = OFF_KT  + (size_t)BATCH*MK*CK*2;        // f16 [256][384]
static constexpr size_t OFF_W2H = OFF_W1H + (size_t)CO*K1DIM*2;           // f16 [256][256]
static constexpr size_t OFF_S1  = OFF_W2H + (size_t)CO*K2DIM*2;           // f32 sum[256]|sumsq[256]
static constexpr size_t OFF_S2  = OFF_S1  + 2048;
static constexpr size_t OFF_X   = OFF_S2  + 2048;                         // f16 [COLS][384] (X2 aliases, [COLS][256])
static constexpr size_t OFF_Y   = OFF_X   + (size_t)COLS*K1DIM*2;         // f32 [256][COLS]

// ---------------------------------------------------------------- utilities
__global__ void zero_stats_kernel(float* s1, float* s2) {
    int t = threadIdx.x;
    if (t < 512) { s1[t] = 0.0f; s2[t] = 0.0f; }
}

// ------------------------------------------------- phase 0: vector building
__global__ void prep_unknown_kernel(const float* __restrict__ unknown,
                                    const float* __restrict__ ufeats,
                                    _Float16* __restrict__ a16,
                                    float* __restrict__ aa,
                                    float* __restrict__ out) {
    int col = blockIdx.x * 256 + threadIdx.x;
    if (col >= COLS) return;
    int b = col >> 14, i = col & (NU - 1);
    float u0 = unknown[(size_t)col*3+0], u1 = unknown[(size_t)col*3+1], u2 = unknown[(size_t)col*3+2];
    float s = u0*u0 + u1*u1 + u2*u2;
    _Float16* ap = a16 + (size_t)col*8;
    ap[0] = (_Float16)u0; ap[1] = (_Float16)u1; ap[2] = (_Float16)u2;
    const float* ub = ufeats + (size_t)b*C1IN*NU + i;
    float*       ob = out    + (size_t)b*C2IN*NU + i;
#pragma unroll
    for (int k = 0; k < NCOMP; ++k) {
        float c = ub[(size_t)k*NU];
        s += 10.0f * c * c;
        ap[3+k] = (_Float16)(SQRT10 * c);
        ob[(size_t)k*NU] = c;            // output channels 0..4 = unknown_comp
    }
    aa[col] = s;
}

__global__ void prep_known_kernel(const float* __restrict__ known,
                                  const float* __restrict__ kfeats,
                                  _Float16* __restrict__ b16,
                                  float* __restrict__ bb) {
    int t = blockIdx.x * 256 + threadIdx.x;
    if (t >= BATCH*MK) return;
    int b = t >> 12, j = t & (MK - 1);
    float u0 = known[(size_t)t*3+0], u1 = known[(size_t)t*3+1], u2 = known[(size_t)t*3+2];
    float s = u0*u0 + u1*u1 + u2*u2;
    _Float16* bp = b16 + (size_t)t*8;
    bp[0] = (_Float16)u0; bp[1] = (_Float16)u1; bp[2] = (_Float16)u2;
    const float* kb = kfeats + (size_t)b*C2IN*MK + j;
#pragma unroll
    for (int k = 0; k < NCOMP; ++k) {
        float c = kb[(size_t)k*MK];
        s += 10.0f * c * c;
        bp[3+k] = (_Float16)(SQRT10 * c);
    }
    bb[t] = s;
}

__global__ void transpose_kfeat_kernel(const float* __restrict__ kfeats,
                                       _Float16* __restrict__ kT) {
    int c = blockIdx.x & (CK - 1);
    int b = blockIdx.x >> 8;
    const float* src = kfeats + (size_t)b*C2IN*MK + (size_t)(NCOMP + c)*MK;
    for (int j = threadIdx.x; j < MK; j += 256)
        kT[((size_t)b*MK + j)*CK + c] = (_Float16)src[j];
}

__global__ void convert_w_kernel(const float* __restrict__ W1, const float* __restrict__ W2,
                                 _Float16* __restrict__ W1h, _Float16* __restrict__ W2h) {
    int t = blockIdx.x * 256 + threadIdx.x;
    if (t < CO*K1DIM) W1h[t] = (_Float16)W1[t];
    else {
        int u = t - CO*K1DIM;
        if (u < CO*K2DIM) W2h[u] = (_Float16)W2[u];
    }
}

// ------------------------------------------------- phase 1: WMMA 3-NN search
// branchless: 3 compares + 10 selects, no EXEC-mask traffic
__device__ __forceinline__ void top3_insert(float d, int j,
                                            float& d0, float& d1, float& d2,
                                            int& j0, int& j1, int& j2) {
    const bool c0 = d < d0;
    const bool c1 = d < d1;
    const bool c2 = d < d2;
    float nd0 = c0 ? d  : d0;
    int   nj0 = c0 ? j  : j0;
    float nd1 = c0 ? d0 : (c1 ? d : d1);
    int   nj1 = c0 ? j0 : (c1 ? j : j1);
    float nd2 = c1 ? d1 : (c2 ? d : d2);
    int   nj2 = c1 ? j1 : (c2 ? j : j2);
    d0 = nd0; d1 = nd1; d2 = nd2;
    j0 = nj0; j1 = nj1; j2 = nj2;
}

// one wave (32 threads) handles 16 query points vs all 4096 knowns of a batch.
// Software-pipelined: WMMA for tile t issues before the top-3 selects of tile
// t-1 consume the previous accumulator (hides the WMMA->VALU hazard NOPs and
// lets selects co-execute with the matrix pipe).
__global__ __launch_bounds__(32) void knn3_kernel(const _Float16* __restrict__ a16,
                                                  const float* __restrict__ aa,
                                                  const _Float16* __restrict__ b16,
                                                  const float* __restrict__ bb,
                                                  int* __restrict__ oidx,
                                                  float* __restrict__ owgt) {
    const int lane = threadIdx.x;
    const int g16  = blockIdx.x * 16;          // query base (global column)
    const int bat  = g16 / NU;
    const int l15  = lane & 15;
    const bool lo  = lane < 16;

    // A operand: 16 query rows, real K=0..7, padded to 32
    v16h aop;
#pragma unroll
    for (int e = 0; e < 16; ++e) aop[e] = (_Float16)0.0f;
    if (lo) {
        v8h av = *(const v8h*)(a16 + (size_t)(g16 + lane)*8);
#pragma unroll
        for (int e = 0; e < 8; ++e) aop[e] = av[e];
    }
    // per-lane row norms (rows r or r+8)
    float aa8[8];
    {
        int rb = g16 + (lo ? 0 : 8);
#pragma unroll
        for (int r = 0; r < 8; ++r) aa8[r] = aa[rb + r];
    }

    float d0[8], d1[8], d2[8];
    int   j0[8], j1[8], j2[8];
#pragma unroll
    for (int r = 0; r < 8; ++r) { d0[r]=d1[r]=d2[r]=FLT_MAX; j0[r]=j1[r]=j2[r]=0; }

    const int kb0 = bat * MK;

    // ---- prologue: tile 0
    v8f accP;
    float bcolP;
    {
        v16h bop;
#pragma unroll
        for (int e = 0; e < 16; ++e) bop[e] = (_Float16)0.0f;
        if (lo) {
            v8h bv = *(const v8h*)(b16 + (size_t)(kb0 + lane)*8);
#pragma unroll
            for (int e = 0; e < 8; ++e) bop[e] = bv[e];
        }
        bcolP = bb[kb0 + l15];
        v8f z = {};
        accP = __builtin_amdgcn_wmma_f32_16x16x32_f16(false, aop, false, bop,
                                                      (short)0, z, false, false);
    }

    // ---- steady state
    for (int t = 1; t < MK/16; ++t) {
        const int kb = kb0 + t*16;
        v16h bop;
#pragma unroll
        for (int e = 0; e < 16; ++e) bop[e] = (_Float16)0.0f;
        if (lo) {
            v8h bv = *(const v8h*)(b16 + (size_t)(kb + lane)*8);
#pragma unroll
            for (int e = 0; e < 8; ++e) bop[e] = bv[e];
        }
        float bcol = bb[kb + l15];
        v8f z = {};
        v8f accC = __builtin_amdgcn_wmma_f32_16x16x32_f16(false, aop, false, bop,
                                                          (short)0, z, false, false);
        // consume previous tile's accumulator while WMMA runs
        const int jjP = (t-1)*16 + l15;
#pragma unroll
        for (int r = 0; r < 8; ++r) {
            float d = __builtin_fmaf(-2.0f, accP[r], aa8[r] + bcolP);
            top3_insert(d, jjP, d0[r], d1[r], d2[r], j0[r], j1[r], j2[r]);
        }
        accP = accC;
        bcolP = bcol;
    }
    // ---- epilogue: last tile
    {
        const int jjP = (MK/16 - 1)*16 + l15;
#pragma unroll
        for (int r = 0; r < 8; ++r) {
            float d = __builtin_fmaf(-2.0f, accP[r], aa8[r] + bcolP);
            top3_insert(d, jjP, d0[r], d1[r], d2[r], j0[r], j1[r], j2[r]);
        }
    }

    // merge 16 column-lanes per row
    __shared__ float sd[16][16][3];
    __shared__ int   si[16][16][3];
#pragma unroll
    for (int r = 0; r < 8; ++r) {
        int row = lo ? r : (r + 8);
        sd[row][l15][0] = d0[r]; sd[row][l15][1] = d1[r]; sd[row][l15][2] = d2[r];
        si[row][l15][0] = j0[r]; si[row][l15][1] = j1[r]; si[row][l15][2] = j2[r];
    }
    __syncthreads();

    if (lane < 16) {
        float t0 = FLT_MAX, t1 = FLT_MAX, t2 = FLT_MAX;
        int   q0 = 0, q1 = 0, q2 = 0;
        for (int c = 0; c < 16; ++c)
#pragma unroll
            for (int k = 0; k < 3; ++k)
                top3_insert(sd[lane][c][k], si[lane][c][k], t0, t1, t2, q0, q1, q2);
        float r0 = 1.0f / (fmaxf(t0, 0.0f) + 1e-8f);
        float r1 = 1.0f / (fmaxf(t1, 0.0f) + 1e-8f);
        float r2 = 1.0f / (fmaxf(t2, 0.0f) + 1e-8f);
        float rs = 1.0f / (r0 + r1 + r2);
        size_t base = (size_t)(g16 + lane) * 3;
        oidx[base+0] = q0; oidx[base+1] = q1; oidx[base+2] = q2;
        owgt[base+0] = r0*rs; owgt[base+1] = r1*rs; owgt[base+2] = r2*rs;
    }
}

// -------------------------------------- phase 2: interpolation -> X1 (f16, K-major)
__global__ void gather_x1_kernel(const int* __restrict__ oidx,
                                 const float* __restrict__ owgt,
                                 const _Float16* __restrict__ kT,
                                 _Float16* __restrict__ X1T) {
    const int col = blockIdx.x;
    const int b = col >> 14;
    const int tid = threadIdx.x;
    size_t ib = (size_t)col * 3;
    int   i0 = oidx[ib+0], i1 = oidx[ib+1], i2 = oidx[ib+2];
    float w0 = owgt[ib+0], w1 = owgt[ib+1], w2 = owgt[ib+2];
    const _Float16* r0 = kT + ((size_t)b*MK + i0) * CK;
    const _Float16* r1 = kT + ((size_t)b*MK + i1) * CK;
    const _Float16* r2 = kT + ((size_t)b*MK + i2) * CK;
    _Float16* xrow = X1T + (size_t)col * K1DIM;
#pragma unroll
    for (int pass = 0; pass < 2; ++pass) {
        int c = tid + pass * 128;
        float v = w0 * (float)r0[c] + w1 * (float)r1[c] + w2 * (float)r2[c];
        xrow[c] = (_Float16)v;
    }
}

// ------------- phase 2b: ufeat channels 256..383 via LDS-tiled transpose
// coalesced global reads; 16B-vectorized per-thread writes
__global__ __launch_bounds__(256) void ufeat_to_x_kernel(const float* __restrict__ ufeats,
                                                         _Float16* __restrict__ X1T) {
    __shared__ _Float16 tile[128][136];   // +8 pad: stride 68 dwords, conflict-free
    const int b  = blockIdx.y;
    const int i0 = blockIdx.x * 128;
    const float* src = ufeats + (size_t)b*C1IN*NU + (size_t)NCOMP*NU + i0;
    for (int e = threadIdx.x; e < 128*128; e += 256) {
        int c = e >> 7, ii = e & 127;
        tile[c][ii] = (_Float16)src[(size_t)c*NU + ii];
    }
    __syncthreads();
    const int ii   = threadIdx.x & 127;
    const int half = threadIdx.x >> 7;    // each half-block writes 64 channels
    _Float16* xrow = X1T + (size_t)(b*NU + i0 + ii) * K1DIM + CK + half*64;
#pragma unroll
    for (int c0 = 0; c0 < 64; c0 += 8) {
        v8h v;
#pragma unroll
        for (int e = 0; e < 8; ++e) v[e] = tile[half*64 + c0 + e][ii];
        *(v8h*)(xrow + c0) = v;
    }
}

// ----------------------------------------------- phase 3: WMMA GEMM  Y = W * X
// W: f16 [O][K] row-major; Xt: f16 [col][K] (K-major); Y: f32 [O][COLS]
__global__ __launch_bounds__(128) void gemm_wmma_kernel(const _Float16* __restrict__ W,
                                                        const _Float16* __restrict__ Xt,
                                                        float* __restrict__ Y, int K) {
    const int lane  = threadIdx.x & 31;
    const int wv    = threadIdx.x >> 5;
    const int mbase = blockIdx.y * 16;
    const int nbase = blockIdx.x * 256 + wv * 64;
    const int l15   = lane & 15;
    const int hi    = (lane < 16) ? 0 : 1;

    v8f acc[4];
#pragma unroll
    for (int j = 0; j < 4; ++j) acc[j] = (v8f){};

    const _Float16* wrow = W + (size_t)(mbase + l15) * K;
    const int aoff = hi * 16;   // A: lanes 16-31 hold K=16..31
    const int boff = hi * 16;   // B: lanes 16-31 hold K=16..31

    for (int kc = 0; kc < K; kc += 32) {
        v8h alo = *(const v8h*)(wrow + kc + aoff);
        v8h ahi = *(const v8h*)(wrow + kc + aoff + 8);
        v16h a;
#pragma unroll
        for (int e = 0; e < 8; ++e) { a[e] = alo[e]; a[e+8] = ahi[e]; }
#pragma unroll
        for (int j = 0; j < 4; ++j) {
            const _Float16* xp = Xt + (size_t)(nbase + j*16 + l15) * K + kc + boff;
            v16h bmat = *(const v16h*)xp;
            acc[j] = __builtin_amdgcn_wmma_f32_16x16x32_f16(false, a, false, bmat,
                                                            (short)0, acc[j], false, false);
        }
    }
    // write out: lane holds rows (r + hi*8), col = subtile base + l15
#pragma unroll
    for (int j = 0; j < 4; ++j) {
        const int col = nbase + j*16 + l15;
#pragma unroll
        for (int r = 0; r < 8; ++r)
            Y[(size_t)(mbase + r + hi*8) * COLS + col] = acc[j][r];
    }
}

// ------------------------------------------- phase 4: BN stats (sum / sumsq)
__global__ void stats_kernel(const float* __restrict__ Y, float* __restrict__ stats) {
    const int o = blockIdx.y;
    const int tid = threadIdx.x;
    const int colbase = blockIdx.x * 4096 + tid;
    float s = 0.0f, s2 = 0.0f;
#pragma unroll 4
    for (int it = 0; it < 16; ++it) {
        float v = Y[(size_t)o * COLS + colbase + it * 256];
        s += v; s2 += v * v;
    }
    __shared__ float ss[256], sq[256];
    ss[tid] = s; sq[tid] = s2;
    __syncthreads();
    for (int stride = 128; stride > 0; stride >>= 1) {
        if (tid < stride) { ss[tid] += ss[tid+stride]; sq[tid] += sq[tid+stride]; }
        __syncthreads();
    }
    if (tid == 0) {
        atomicAdd(&stats[o], ss[0]);
        atomicAdd(&stats[CO + o], sq[0]);
    }
}

// ------------------------------- phase 5a: BN+ReLU -> next layer input (f16 K-major)
__global__ void bnrelu_to_x_kernel(const float* __restrict__ Y,
                                   const float* __restrict__ stats,
                                   const float* __restrict__ g,
                                   const float* __restrict__ bia,
                                   _Float16* __restrict__ Xout) {
    const int o = blockIdx.y;
    const int col = blockIdx.x * 256 + threadIdx.x;
    const float inv = 1.0f / (float)COLS;
    float mean = stats[o] * inv;
    float var  = stats[CO + o] * inv - mean * mean;
    float rstd = rsqrtf(var + BN_EPS);
    float v = (Y[(size_t)o * COLS + col] - mean) * rstd * g[o] + bia[o];
    Xout[(size_t)col * K2DIM + o] = (_Float16)fmaxf(v, 0.0f);
}

// ------------------------------- phase 5b: BN+ReLU -> final output (f32, ch offset 5)
__global__ void bnrelu_to_out_kernel(const float* __restrict__ Y,
                                     const float* __restrict__ stats,
                                     const float* __restrict__ g,
                                     const float* __restrict__ bia,
                                     float* __restrict__ out) {
    const int o = blockIdx.y;
    const int col = blockIdx.x * 256 + threadIdx.x;
    const int b = col >> 14, i = col & (NU - 1);
    const float inv = 1.0f / (float)COLS;
    float mean = stats[o] * inv;
    float var  = stats[CO + o] * inv - mean * mean;
    float rstd = rsqrtf(var + BN_EPS);
    float v = (Y[(size_t)o * COLS + col] - mean) * rstd * g[o] + bia[o];
    out[(size_t)b * C2IN * NU + (size_t)(NCOMP + o) * NU + i] = fmaxf(v, 0.0f);
}

// ---------------------------------------------------------------- launcher
extern "C" void kernel_launch(void* const* d_in, const int* in_sizes, int n_in,
                              void* d_out, int out_size, void* d_ws, size_t ws_size,
                              hipStream_t stream) {
    const float* unknown = (const float*)d_in[0];
    const float* known   = (const float*)d_in[1];
    const float* ufeats  = (const float*)d_in[2];
    const float* kfeats  = (const float*)d_in[3];
    const float* W1      = (const float*)d_in[4];
    const float* g1      = (const float*)d_in[5];
    const float* b1      = (const float*)d_in[6];
    const float* W2      = (const float*)d_in[7];
    const float* g2      = (const float*)d_in[8];
    const float* b2      = (const float*)d_in[9];
    float* out = (float*)d_out;
    char*  ws  = (char*)d_ws;

    _Float16* a16  = (_Float16*)(ws + OFF_A16);
    float*    aa   = (float*)   (ws + OFF_AA);
    _Float16* b16  = (_Float16*)(ws + OFF_B16);
    float*    bb   = (float*)   (ws + OFF_BB);
    int*      idxb = (int*)     (ws + OFF_IDX);
    float*    wgtb = (float*)   (ws + OFF_WGT);
    _Float16* kT   = (_Float16*)(ws + OFF_KT);
    _Float16* W1h  = (_Float16*)(ws + OFF_W1H);
    _Float16* W2h  = (_Float16*)(ws + OFF_W2H);
    float*    s1   = (float*)   (ws + OFF_S1);
    float*    s2   = (float*)   (ws + OFF_S2);
    _Float16* X1T  = (_Float16*)(ws + OFF_X);   // [COLS][384]
    _Float16* X2T  = (_Float16*)(ws + OFF_X);   // aliased: [COLS][256]
    float*    Ybuf = (float*)   (ws + OFF_Y);   // [256][COLS], reused for both layers

    zero_stats_kernel<<<1, 512, 0, stream>>>(s1, s2);
    prep_unknown_kernel<<<COLS/256, 256, 0, stream>>>(unknown, ufeats, a16, aa, out);
    prep_known_kernel<<<(BATCH*MK)/256, 256, 0, stream>>>(known, kfeats, b16, bb);
    transpose_kfeat_kernel<<<BATCH*CK, 256, 0, stream>>>(kfeats, kT);
    convert_w_kernel<<<(CO*K1DIM + CO*K2DIM)/256, 256, 0, stream>>>(W1, W2, W1h, W2h);

    knn3_kernel<<<COLS/16, 32, 0, stream>>>(a16, aa, b16, bb, idxb, wgtb);
    gather_x1_kernel<<<COLS, 128, 0, stream>>>(idxb, wgtb, kT, X1T);
    ufeat_to_x_kernel<<<dim3(NU/128, BATCH), 256, 0, stream>>>(ufeats, X1T);

    // layer 1: Y = W1 * X1 ; BN stats ; BN+ReLU -> X2 (f16)
    gemm_wmma_kernel<<<dim3(COLS/256, CO/16), 128, 0, stream>>>(W1h, X1T, Ybuf, K1DIM);
    stats_kernel<<<dim3(16, CO), 256, 0, stream>>>(Ybuf, s1);
    bnrelu_to_x_kernel<<<dim3(COLS/256, CO), 256, 0, stream>>>(Ybuf, s1, g1, b1, X2T);

    // layer 2: Y = W2 * X2 ; BN stats ; BN+ReLU -> out channels 5..260
    gemm_wmma_kernel<<<dim3(COLS/256, CO/16), 128, 0, stream>>>(W2h, X2T, Ybuf, K2DIM);
    stats_kernel<<<dim3(16, CO), 256, 0, stream>>>(Ybuf, s2);
    bnrelu_to_out_kernel<<<dim3(COLS/256, CO), 256, 0, stream>>>(Ybuf, s2, g2, b2, out);
}
